// BioEncoder_36017595744530
// MI455X (gfx1250) — compile-verified, matrix operand order checked by reading, not run on hardware
//
#include <hip/hip_runtime.h>

// ---------------- problem constants (match reference) ----------------
#define NNODES 100000
#define NEDGES 400000
#define OUTD   256
#define HEADS  4
#define HDIM   64
#define NGRAPH 2048
#define BCL    2048
#define CDIM   954
#define CPAD   960
#define DDIM   75
#define DPAD   96
#define BNEPS  1e-5f
#define BM     128              // block M tile (TDM-staged in LDS)
#define LDSROW 40               // LDS row stride in halves (32 data + 8 pad -> 80B, bank-conflict free)

typedef __attribute__((ext_vector_type(16))) _Float16 v16h;
typedef __attribute__((ext_vector_type(8)))  _Float16 v8h;
typedef __attribute__((ext_vector_type(8)))  float    v8f;
typedef __attribute__((ext_vector_type(4)))  unsigned u32x4;
typedef __attribute__((ext_vector_type(8)))  unsigned u32x8;

// ---------------- small utility kernels ----------------
__global__ void fill_f32(float* __restrict__ p, float v, long long n) {
    long long i = (long long)blockIdx.x * blockDim.x + threadIdx.x;
    if (i < n) p[i] = v;
}

__global__ void convert_pad_f16(const float* __restrict__ X, _Float16* __restrict__ Y,
                                int M, int K, int Kpad) {
    long long idx = (long long)blockIdx.x * blockDim.x + threadIdx.x;
    long long total = (long long)M * Kpad;
    if (idx >= total) return;
    int r = (int)(idx / Kpad), c = (int)(idx % Kpad);
    Y[idx] = (c < K) ? (_Float16)X[(long long)r * K + c] : (_Float16)0.0f;
}

// W [K, Nc] f32 row-major -> Wt [Nc, Kpad] f16 (zero pad K)
__global__ void transpose_convert_f16(const float* __restrict__ W, _Float16* __restrict__ Wt,
                                      int K, int Nc, int Kpad) {
    long long idx = (long long)blockIdx.x * blockDim.x + threadIdx.x;
    long long total = (long long)Nc * Kpad;
    if (idx >= total) return;
    int n = (int)(idx / Kpad), kc = (int)(idx % Kpad);
    Wt[idx] = (kc < K) ? (_Float16)W[(long long)kc * Nc + n] : (_Float16)0.0f;
}

// ---------------- TDM-staged WMMA GEMM ----------------
// C[M,Nc] = act(A16[M,Kpad] * Wt16^T + bias)
// Block: 128 threads (4 waves), tile 128M x 64N. Per K-step of 32:
//   - wave 0 issues TENSOR_LOAD_TO_LDS for the 128x32 f16 A tile (padded to 80B rows),
//     waits TENSORcnt, workgroup barrier publishes it.
//   - each wave: 1 B fragment (2 global b128) + 8 A fragments from LDS (ds_load_b128)
//     -> 8 v_wmma_f32_16x16x32_f16.
// act: 0=none 1=relu 2=tanh
__global__ __launch_bounds__(128)
void wmma_gemm(const _Float16* __restrict__ A, const _Float16* __restrict__ Bt,
               const float* __restrict__ bias, float* __restrict__ C,
               int M, int Kpad, int Nc, int act) {
    __shared__ _Float16 atile[BM * LDSROW];

    const int wave = threadIdx.x >> 5;
    const int lane = threadIdx.x & 31;
    const int g    = lane >> 4;          // lane half-group -> K mid-chunk select
    const int l16  = lane & 15;
    const int m0   = blockIdx.x * BM;
    const int col  = blockIdx.y * 64 + wave * 16 + l16;

    const _Float16* Bp = Bt + (long long)col * Kpad + g * 8;
    const unsigned ldsoff  = (unsigned)(size_t)(&atile[0]);   // low 32 bits = LDS byte offset
    const unsigned remrows = (unsigned)(M - m0);              // TDM OOB rows read as zero

    v8f acc[BM / 16];
#pragma unroll
    for (int i = 0; i < BM / 16; ++i) acc[i] = (v8f){};

    for (int kk = 0; kk < Kpad; kk += 32) {
        __syncthreads();                 // previous tile fully consumed
        if (wave == 0) {
            unsigned long long ga =
                (unsigned long long)(size_t)A + ((long long)m0 * Kpad + kk) * 2ll;
            u32x4 g0;
            u32x8 g1;
            // D# group0: count=1 | lds_addr | global_addr[56:0] | type=2
            g0[0] = 1u;
            g0[1] = ldsoff;
            g0[2] = (unsigned)ga;
            g0[3] = ((unsigned)(ga >> 32) & 0x1FFFFFFu) | (2u << 30);
            // D# group1: data_size=2B, pad_enable, pad_interval=16 DW, pad_amount=4 DW
            g0 = g0;  // (keep g0 live)
            g1[0] = 0x06D10000u;
            g1[1] = ((unsigned)Kpad) << 16;                     // tensor_dim0[15:0]
            g1[2] = (((unsigned)Kpad) >> 16) | (remrows << 16); // tensor_dim0[31:16] | tensor_dim1[15:0]
            g1[3] = (remrows >> 16) | (32u << 16);              // tensor_dim1[31:16] | tile_dim0=32
            g1[4] = (unsigned)BM;                               // tile_dim1=128, tile_dim2=0
            g1[5] = (unsigned)Kpad;                             // tensor_dim0_stride[31:0]
            g1[6] = 0u;
            g1[7] = 0u;
            asm volatile("tensor_load_to_lds %0, %1" :: "s"(g0), "s"(g1) : "memory");
            __builtin_amdgcn_s_wait_tensorcnt(0);
        }
        __syncthreads();                 // tile visible to all waves

        // B fragment: halves 0..7 -> K=g*8+0..7 ; halves 8..15 -> K=g*8+16..23
        v8h b0 = *(const v8h*)(Bp + kk);
        v8h b1 = *(const v8h*)(Bp + kk + 16);
        v16h bv;
#pragma unroll
        for (int i = 0; i < 8; ++i) { bv[i] = b0[i]; bv[i + 8] = b1[i]; }

#pragma unroll
        for (int mi = 0; mi < BM / 16; ++mi) {
            const _Float16* ap = atile + (mi * 16 + l16) * LDSROW + g * 8;
            v8h a0 = *(const v8h*)ap;          // ds_load_b128
            v8h a1 = *(const v8h*)(ap + 16);   // ds_load_b128
            v16h av;
#pragma unroll
            for (int i = 0; i < 8; ++i) { av[i] = a0[i]; av[i + 8] = a1[i]; }
            acc[mi] = __builtin_amdgcn_wmma_f32_16x16x32_f16(false, av, false, bv,
                                                             (short)0, acc[mi], false, false);
        }
    }

    const float badd = bias ? bias[col] : 0.0f;
#pragma unroll
    for (int mi = 0; mi < BM / 16; ++mi) {
#pragma unroll
        for (int r = 0; r < 8; ++r) {
            int row = m0 + mi * 16 + g * 8 + r;  // C layout: lanes 0-15 rows 0..7, 16-31 rows 8..15
            if (row < M) {
                float x = acc[mi][r] + badd;
                if (act == 1) x = fmaxf(x, 0.0f);
                else if (act == 2) x = tanhf(x);
                C[(long long)row * Nc + col] = x;
            }
        }
    }
}

// ---------------- scatter-softmax attention ----------------
__device__ inline void atomicMaxF(float* addr, float value) {
    if (value >= 0.0f) atomicMax((int*)addr, __float_as_int(value));
    else               atomicMin((unsigned int*)addr, __float_as_uint(value));
}

__global__ __launch_bounds__(256)
void edge_logits_max(const int* __restrict__ src, const int* __restrict__ dst,
                     const float* __restrict__ q, const float* __restrict__ k,
                     float* __restrict__ logits, float* __restrict__ mbuf) {
    int idx = blockIdx.x * blockDim.x + threadIdx.x;
    if (idx >= NEDGES * HEADS) return;
    int e = idx >> 2, h = idx & 3;
    int s = src[e], d = dst[e];
    const float4* qp = (const float4*)(q + (long long)d * OUTD + h * HDIM);
    const float4* kp = (const float4*)(k + (long long)s * OUTD + h * HDIM);
    float acc = 0.0f;
#pragma unroll
    for (int i = 0; i < HDIM / 4; ++i) {
        float4 a = qp[i], b = kp[i];
        acc += a.x * b.x + a.y * b.y + a.z * b.z + a.w * b.w;
    }
    float lg = acc * 0.125f;           // 1/sqrt(64)
    logits[idx] = lg;
    atomicMaxF(&mbuf[(long long)d * HEADS + h], lg);
}

__global__ __launch_bounds__(256)
void edge_expsum(const int* __restrict__ dst, float* __restrict__ logits,
                 const float* __restrict__ mbuf, float* __restrict__ sbuf) {
    int idx = blockIdx.x * blockDim.x + threadIdx.x;
    if (idx >= NEDGES * HEADS) return;
    int e = idx >> 2, h = idx & 3;
    int d = dst[e];
    float m = mbuf[(long long)d * HEADS + h];
    if (!(m > -1e37f && m < 1e37f)) m = 0.0f;   // isfinite fixup (matches reference)
    float ev = __expf(logits[idx] - m);
    logits[idx] = ev;
    atomicAdd(&sbuf[(long long)d * HEADS + h], ev);
}

__global__ __launch_bounds__(256)
void edge_scatter(const int* __restrict__ src, const int* __restrict__ dst,
                  const float* __restrict__ logits, const float* __restrict__ sbuf,
                  const float* __restrict__ v, float* __restrict__ out) {
    long long idx = (long long)blockIdx.x * blockDim.x + threadIdx.x;
    if (idx >= (long long)NEDGES * OUTD) return;
    int e = (int)(idx >> 8), c = (int)(idx & 255), h = c >> 6;
    int s = src[e], d = dst[e];
    float alpha = logits[(long long)e * HEADS + h] /
                  (sbuf[(long long)d * HEADS + h] + 1e-16f);
    atomicAdd(&out[(long long)d * OUTD + c], alpha * v[(long long)s * OUTD + c]);
}

// ---------------- BatchNorm (training stats, biased var) ----------------
__global__ __launch_bounds__(256)
void bn_partial(const float* __restrict__ X, int n, int relu,
                float* __restrict__ sum, float* __restrict__ sumsq) {
    int c = threadIdx.x;               // 256 channels, coalesced column reads
    float s = 0.0f, s2 = 0.0f;
    for (int i = blockIdx.x; i < n; i += gridDim.x) {
        float x = X[(long long)i * OUTD + c];
        if (relu) x = fmaxf(x, 0.0f);
        s += x; s2 += x * x;
    }
    atomicAdd(&sum[c], s);
    atomicAdd(&sumsq[c], s2);
}

__global__ void bn_finalize(const float* __restrict__ sum, const float* __restrict__ sumsq,
                            float n, float* __restrict__ mean, float* __restrict__ invstd) {
    int c = threadIdx.x;
    float mu = sum[c] / n;
    float var = sumsq[c] / n - mu * mu;
    mean[c] = mu;
    invstd[c] = rsqrtf(var + BNEPS);
}

__global__ __launch_bounds__(256)
void bn_apply(const float* __restrict__ X, const float* __restrict__ mean,
              const float* __restrict__ invstd, const float* __restrict__ gamma,
              const float* __restrict__ beta, const float* __restrict__ resid,
              float* __restrict__ Y, int n, int relu) {
    long long idx = (long long)blockIdx.x * blockDim.x + threadIdx.x;
    if (idx >= (long long)n * OUTD) return;
    int c = (int)(idx & 255);
    float x = X[idx];
    if (relu) x = fmaxf(x, 0.0f);
    float y = (x - mean[c]) * invstd[c] * gamma[c] + beta[c];
    if (resid) y += resid[idx];
    Y[idx] = y;
}

// ---------------- pooling + elementwise ----------------
__global__ void pool_count(const int* __restrict__ ib, float* __restrict__ counts) {
    int i = blockIdx.x * blockDim.x + threadIdx.x;
    if (i < NNODES) atomicAdd(&counts[ib[i]], 1.0f);
}

__global__ void pool_sum(const float* __restrict__ h, const int* __restrict__ ib,
                         float* __restrict__ pooled) {
    long long idx = (long long)blockIdx.x * blockDim.x + threadIdx.x;
    if (idx >= (long long)NNODES * OUTD) return;
    int i = (int)(idx >> 8), c = (int)(idx & 255);
    atomicAdd(&pooled[(long long)ib[i] * OUTD + c], h[idx]);
}

__global__ void pool_div(const float* __restrict__ pooled, const float* __restrict__ counts,
                         float* __restrict__ out) {
    long long idx = (long long)blockIdx.x * blockDim.x + threadIdx.x;
    if (idx >= (long long)NGRAPH * OUTD) return;
    out[idx] = pooled[idx] / fmaxf(counts[idx >> 8], 1.0f);
}

__global__ void add_vec(const float* __restrict__ a, const float* __restrict__ b,
                        float* __restrict__ out, long long n) {
    long long i = (long long)blockIdx.x * blockDim.x + threadIdx.x;
    if (i < n) out[i] = a[i] + b[i];
}

// ---------------- orchestration ----------------
extern "C" void kernel_launch(void* const* d_in, const int* in_sizes, int n_in,
                              void* d_out, int out_size, void* d_ws, size_t ws_size,
                              hipStream_t stream) {
    const float* drug_x  = (const float*)d_in[0];
    const int*   adj     = (const int*)d_in[1];
    const int*   src     = adj;
    const int*   dst     = adj + NEDGES;
    const int*   ibatch  = (const int*)d_in[2];
    const float* cline_x = (const float*)d_in[3];
    const float *Wq1=(const float*)d_in[4],  *bq1=(const float*)d_in[5];
    const float *Wk1=(const float*)d_in[6],  *bk1=(const float*)d_in[7];
    const float *Wv1=(const float*)d_in[8],  *bv1=(const float*)d_in[9];
    const float *g1 =(const float*)d_in[10], *be1=(const float*)d_in[11];
    const float *Wq2=(const float*)d_in[12], *bq2=(const float*)d_in[13];
    const float *Wk2=(const float*)d_in[14], *bk2=(const float*)d_in[15];
    const float *Wv2=(const float*)d_in[16], *bv2=(const float*)d_in[17];
    const float *g2 =(const float*)d_in[18], *be2=(const float*)d_in[19];
    const float *Wc1=(const float*)d_in[20], *bc1=(const float*)d_in[21];
    const float *gc =(const float*)d_in[22], *bec=(const float*)d_in[23];
    const float *Wc2=(const float*)d_in[24], *bc2=(const float*)d_in[25];

    // ---- workspace carve (≈480 MB) ----
    char* base = (char*)d_ws;
    size_t off = 0;
    auto carve = [&](size_t bytes) -> void* {
        void* r = base + off;
        off = (off + bytes + 255) & ~(size_t)255;
        return r;
    };
    const long long NOUT = (long long)NNODES * OUTD;
    float* q    = (float*)carve(NOUT * 4);               // also reused as attention accumulator
    float* kbuf = (float*)carve(NOUT * 4);
    float* v    = (float*)carve(NOUT * 4);               // also reused as final node features
    float* h1   = (float*)carve(NOUT * 4);
    float* logits = (float*)carve((long long)NEDGES * HEADS * 4);
    float* mbuf = (float*)carve((long long)NNODES * HEADS * 4);
    float* sbuf = (float*)carve((long long)NNODES * HEADS * 4);
    float* bsum = (float*)carve(OUTD * 4);
    float* bssq = (float*)carve(OUTD * 4);
    float* bmean = (float*)carve(OUTD * 4);
    float* binv  = (float*)carve(OUTD * 4);
    _Float16* x16  = (_Float16*)carve(NOUT * 2);         // GEMM A operand (layer1 uses front N*96)
    _Float16* wt16 = (_Float16*)carve((long long)CPAD * OUTD * 2);
    float* pooled = (float*)carve((long long)NGRAPH * OUTD * 4);
    float* counts = (float*)carve((long long)NGRAPH * 4);
    float* c1  = (float*)carve((long long)BCL * OUTD * 4);
    float* cbn = (float*)carve((long long)BCL * OUTD * 4);
    float* c2  = (float*)carve((long long)BCL * OUTD * 4);
    _Float16* c16 = (_Float16*)carve((long long)BCL * CPAD * 2);

    auto blocks = [](long long n) { return (unsigned)((n + 255) / 256); };

    auto gemm = [&](const _Float16* A, const float* W, const float* bias, float* C,
                    int M, int K, int Kpad, int Nc, int act) {
        long long tn = (long long)Nc * Kpad;
        transpose_convert_f16<<<blocks(tn), 256, 0, stream>>>(W, wt16, K, Nc, Kpad);
        dim3 grid((M + BM - 1) / BM, Nc / 64);
        wmma_gemm<<<grid, 128, 0, stream>>>(A, wt16, bias, C, M, Kpad, Nc, act);
    };

    auto tlayer = [&](const _Float16* A16, int K, int Kpad,
                      const float* Wq, const float* bq, const float* Wk, const float* bk,
                      const float* Wv, const float* bvp,
                      const float* gamma, const float* beta,
                      const float* resid, float* hout) {
        gemm(A16, Wq, bq, q,    NNODES, K, Kpad, OUTD, 0);
        gemm(A16, Wk, bk, kbuf, NNODES, K, Kpad, OUTD, 0);
        gemm(A16, Wv, bvp, v,   NNODES, K, Kpad, OUTD, 0);
        long long nh = (long long)NNODES * HEADS;
        fill_f32<<<blocks(nh), 256, 0, stream>>>(mbuf, -__builtin_inff(), nh);
        fill_f32<<<blocks(nh), 256, 0, stream>>>(sbuf, 0.0f, nh);
        long long eh = (long long)NEDGES * HEADS;
        edge_logits_max<<<blocks(eh), 256, 0, stream>>>(src, dst, q, kbuf, logits, mbuf);
        edge_expsum<<<blocks(eh), 256, 0, stream>>>(dst, logits, mbuf, sbuf);
        fill_f32<<<blocks(NOUT), 256, 0, stream>>>(q, 0.0f, NOUT);   // q reused as accumulator
        edge_scatter<<<blocks((long long)NEDGES * OUTD), 256, 0, stream>>>(src, dst, logits, sbuf, v, q);
        fill_f32<<<1, 256, 0, stream>>>(bsum, 0.0f, OUTD);
        fill_f32<<<1, 256, 0, stream>>>(bssq, 0.0f, OUTD);
        bn_partial<<<256, 256, 0, stream>>>(q, NNODES, 1, bsum, bssq);
        bn_finalize<<<1, 256, 0, stream>>>(bsum, bssq, (float)NNODES, bmean, binv);
        bn_apply<<<blocks(NOUT), 256, 0, stream>>>(q, bmean, binv, gamma, beta, resid,
                                                   hout, NNODES, 1);
    };

    // ---- drug branch ----
    convert_pad_f16<<<blocks((long long)NNODES * DPAD), 256, 0, stream>>>(drug_x, x16, NNODES, DDIM, DPAD);
    tlayer(x16, DDIM, DPAD, Wq1, bq1, Wk1, bk1, Wv1, bv1, g1, be1, nullptr, h1);
    convert_pad_f16<<<blocks(NOUT), 256, 0, stream>>>(h1, x16, NNODES, OUTD, OUTD);
    tlayer(x16, OUTD, OUTD, Wq2, bq2, Wk2, bk2, Wv2, bv2, g2, be2, h1, v);  // v = final node features

    // ---- global mean pool ----
    long long gp = (long long)NGRAPH * OUTD;
    fill_f32<<<blocks(gp), 256, 0, stream>>>(pooled, 0.0f, gp);
    fill_f32<<<blocks(NGRAPH), 256, 0, stream>>>(counts, 0.0f, NGRAPH);
    pool_count<<<blocks(NNODES), 256, 0, stream>>>(ibatch, counts);
    pool_sum<<<blocks(NOUT), 256, 0, stream>>>(v, ibatch, pooled);
    pool_div<<<blocks(gp), 256, 0, stream>>>(pooled, counts, (float*)d_out);

    // ---- cline branch ----
    convert_pad_f16<<<blocks((long long)BCL * CPAD), 256, 0, stream>>>(cline_x, c16, BCL, CDIM, CPAD);
    gemm(c16, Wc1, bc1, c1, BCL, CDIM, CPAD, OUTD, 2);                 // tanh
    fill_f32<<<1, 256, 0, stream>>>(bsum, 0.0f, OUTD);
    fill_f32<<<1, 256, 0, stream>>>(bssq, 0.0f, OUTD);
    bn_partial<<<256, 256, 0, stream>>>(c1, BCL, 0, bsum, bssq);
    bn_finalize<<<1, 256, 0, stream>>>(bsum, bssq, (float)BCL, bmean, binv);
    bn_apply<<<blocks((long long)BCL * OUTD), 256, 0, stream>>>(c1, bmean, binv, gc, bec,
                                                               nullptr, cbn, BCL, 0);
    convert_pad_f16<<<blocks((long long)BCL * OUTD), 256, 0, stream>>>(cbn, c16, BCL, OUTD, OUTD);
    gemm(c16, Wc2, bc2, c2, BCL, OUTD, OUTD, OUTD, 1);                 // relu
    add_vec<<<blocks((long long)BCL * OUTD), 256, 0, stream>>>(c1, c2, (float*)d_out + gp,
                                                              (long long)BCL * OUTD);
}
